// MultiHeadHGNN_conv_57801669869836
// MI455X (gfx1250) — matrix operand order repaired and checked by reading, not compiled, not run on hardware
//
#include <hip/hip_runtime.h>
#include <hip/hip_bf16.h>

typedef __attribute__((ext_vector_type(16))) _Float16 v16h;
typedef __attribute__((ext_vector_type(8)))  float    v8f;

// Problem constants (match reference)
constexpr int Bc   = 2;
constexpr int Sc   = 4096;
constexpr int Dc   = 256;     // D_IN == D_OUT
constexpr int Hc   = 4;
constexpr int HDc  = 64;
constexpr int BSc  = Bc * Sc; // 8192

// -------- Workspace layout (in halfs) --------
// W16   : 4 * 256*256            = 262144   (q,k,v,o ; q pre-scaled by 1/8)
// Q16   : [B,H,S,HD]             = 2097152
// K16   : [B,H,S,HD]             = 2097152
// Vt16  : [B,H,HD,S]             = 2097152
// C16   : [B*S, 256]             = 2097152
// total = 8650752 halfs = 17,301,504 bytes
constexpr size_t OFF_W16  = 0;
constexpr size_t OFF_Q16  = 262144;
constexpr size_t OFF_K16  = OFF_Q16 + (size_t)2097152;
constexpr size_t OFF_VT16 = OFF_K16 + (size_t)2097152;
constexpr size_t OFF_C16  = OFF_VT16 + (size_t)2097152;

// ---------------------------------------------------------------------------
// Kernel 1: fp32 -> f16 weight conversion; Wq gets the 1/sqrt(HD)=0.125 fold.
// Layout kept as [out_ch][in_ch] so B-fragments gather contiguous halfs.
// ---------------------------------------------------------------------------
__global__ void wconv_kernel(const float* __restrict__ Wq,
                             const float* __restrict__ Wk,
                             const float* __restrict__ Wv,
                             const float* __restrict__ Wo,
                             _Float16* __restrict__ w16)
{
    int i = blockIdx.x * blockDim.x + threadIdx.x;   // 0 .. 4*65536-1
    int w = i >> 16;
    int e = i & 65535;
    const float* src = (w == 0) ? Wq : (w == 1) ? Wk : (w == 2) ? Wv : Wo;
    float v = src[e];
    if (w == 0) v *= 0.125f;                         // fold score scale into Q
    w16[i] = (_Float16)v;
}

// ---------------------------------------------------------------------------
// Kernel 2: QKV projection. One wave per 16x16 output tile.
//   A = x (fp32 -> f16 on the fly), B = W16 (contiguous 16-half runs).
//   Q/K stored [B,H,S,HD]; V stored transposed [B,H,HD,S].
// ---------------------------------------------------------------------------
__global__ void qkv_proj_kernel(const float* __restrict__ x,
                                const _Float16* __restrict__ w16,
                                _Float16* __restrict__ q16,
                                _Float16* __restrict__ k16,
                                _Float16* __restrict__ vt16)
{
    const int lane = threadIdx.x;
    const int half = lane >> 4;     // 0/1: which 16-lane group
    const int lm   = lane & 15;
    const int mt   = blockIdx.x;    // row tile over B*S (512)
    const int nt   = blockIdx.y;    // col tile over D   (16)
    const int w    = blockIdx.z;    // 0=q 1=k 2=v
    const _Float16* wt = w16 + (size_t)w * Dc * Dc;

    v8f acc = {};
    for (int k0 = 0; k0 < Dc; k0 += 32) {
        // A fragment: row m = mt*16+lm, K = k0 + half*16 + e  (contiguous run)
        v16h a;
        const float* xr = x + (size_t)(mt * 16 + lm) * Dc + k0 + half * 16;
#pragma unroll
        for (int e = 0; e < 8; ++e) {
            a[e]     = (_Float16)xr[e];
            a[e + 8] = (_Float16)xr[8 + e];
        }
        // B fragment: col n = nt*16+lm, K = k0 + half*16 + e  (contiguous run)
        v16h b;
        const _Float16* wr = wt + (size_t)(nt * 16 + lm) * Dc + k0 + half * 16;
#pragma unroll
        for (int e = 0; e < 16; ++e) b[e] = wr[e];

        acc = __builtin_amdgcn_wmma_f32_16x16x32_f16(false, a, false, b,
                                                     (short)0, acc, false, false);
    }

    // C/D layout: lane holds col n=lm; vgpr r holds row m = r + half*8
#pragma unroll
    for (int r = 0; r < 8; ++r) {
        int row = mt * 16 + r + half * 8;     // over B*S
        int b_  = row >> 12;                  // / 4096
        int s_  = row & (Sc - 1);
        int c   = nt * 16 + lm;               // 0..255
        int h_  = c >> 6;
        int d_  = c & (HDc - 1);
        size_t bh = (size_t)(b_ * Hc + h_);
        _Float16 v = (_Float16)acc[r];
        if (w == 2)      vt16[(bh * HDc + d_) * Sc + s_] = v;   // [B,H,HD,S]
        else if (w == 1) k16 [(bh * Sc + s_) * HDc + d_] = v;   // [B,H,S,HD]
        else             q16 [(bh * Sc + s_) * HDc + d_] = v;
    }
}

// ---------------------------------------------------------------------------
// Kernel 3: flash attention. One wave per (b,h, 16 q-rows).
//   Streams K in 32-wide blocks; online softmax; O accumulators resident.
//   scores already include 1/8 (folded into Wq); G applied multiplicatively.
// ---------------------------------------------------------------------------
__global__ void flash_attn_kernel(const _Float16* __restrict__ Q,
                                  const _Float16* __restrict__ K,
                                  const _Float16* __restrict__ Vt,
                                  const float* __restrict__ G,
                                  _Float16* __restrict__ C)
{
    __shared__ _Float16 pt[16 * 34];     // P transpose staging (padded rows)

    const int lane = threadIdx.x;
    const int half = lane >> 4;
    const int lm   = lane & 15;
    const int tile = blockIdx.x;                 // 0..2047
    const int qt   = tile & (Sc / 16 - 1);       // 256 q tiles
    const int bh   = tile >> 8;                  // 0..7
    const int qbase = qt * 16;

    const _Float16* Qb = Q  + (size_t)bh * Sc * HDc;
    const _Float16* Kb = K  + (size_t)bh * Sc * HDc;
    const _Float16* Vb = Vt + (size_t)bh * HDc * Sc;

    // Resident Q A-fragments (16 x 64 -> two 16x32 fragments)
    v16h qa[2];
#pragma unroll
    for (int f = 0; f < 2; ++f) {
        const _Float16* qr = Qb + (size_t)(qbase + lm) * HDc + f * 32 + half * 16;
#pragma unroll
        for (int e = 0; e < 8; ++e) {
            qa[f][e]     = qr[e];
            qa[f][e + 8] = qr[8 + e];
        }
    }

    v8f o0 = {}, o1 = {}, o2 = {}, o3 = {};      // O accum: 16 x 64
    float rmax[8], rsum[8];
#pragma unroll
    for (int r = 0; r < 8; ++r) { rmax[r] = -1e30f; rsum[r] = 0.0f; }

    for (int kb = 0; kb < Sc; kb += 32) {
        // ---- scores: two 16x16 tiles (k-position groups) -------------------
        v8f sc0 = {}, sc1 = {};
#pragma unroll
        for (int f = 0; f < 2; ++f) {
            // B = K^T fragment: col = kpos, e -> d = f*32 + half*16 + e (contiguous)
            v16h b0, b1;
            const _Float16* kr0 = Kb + (size_t)(kb + lm)      * HDc + f * 32 + half * 16;
            const _Float16* kr1 = Kb + (size_t)(kb + 16 + lm) * HDc + f * 32 + half * 16;
#pragma unroll
            for (int e = 0; e < 16; ++e) { b0[e] = kr0[e]; b1[e] = kr1[e]; }
            sc0 = __builtin_amdgcn_wmma_f32_16x16x32_f16(false, qa[f], false, b0,
                                                         (short)0, sc0, false, false);
            sc1 = __builtin_amdgcn_wmma_f32_16x16x32_f16(false, qa[f], false, b1,
                                                         (short)0, sc1, false, false);
        }

        // ---- G mask + online softmax (row stats replicated over 16 lanes) --
        float p0[8], p1[8];
#pragma unroll
        for (int r = 0; r < 8; ++r) {
            int qrow = qbase + r + half * 8;
            float g0 = G[(size_t)qrow * Sc + kb + lm];
            float g1 = G[(size_t)qrow * Sc + kb + 16 + lm];
            float s0 = sc0[r] * g0;
            float s1 = sc1[r] * g1;

            float t = fmaxf(s0, s1);
#pragma unroll
            for (int m = 1; m < 16; m <<= 1) t = fmaxf(t, __shfl_xor(t, m));
            float nm   = fmaxf(rmax[r], t);
            float corr = __expf(rmax[r] - nm);
            rmax[r] = nm;

            float e0 = __expf(s0 - nm);
            float e1 = __expf(s1 - nm);
            float ps = e0 + e1;
#pragma unroll
            for (int m = 1; m < 16; m <<= 1) ps += __shfl_xor(ps, m);
            rsum[r] = rsum[r] * corr + ps;

            o0[r] = o0[r] * corr;
            o1[r] = o1[r] * corr;
            o2[r] = o2[r] * corr;
            o3[r] = o3[r] * corr;
            p0[r] = e0;
            p1[r] = e1;
        }

        // ---- transpose P (C-layout -> A-layout) through LDS ----------------
        __syncthreads();
#pragma unroll
        for (int r = 0; r < 8; ++r) {
            int row = r + half * 8;
            pt[row * 34 + lm]      = (_Float16)p0[r];
            pt[row * 34 + 16 + lm] = (_Float16)p1[r];
        }
        __syncthreads();
        v16h pa;
#pragma unroll
        for (int e = 0; e < 8; ++e) {
            pa[e]     = pt[lm * 34 + half * 16 + e];
            pa[e + 8] = pt[lm * 34 + half * 16 + 8 + e];
        }

        // ---- O += P @ V : four HD column tiles -----------------------------
        {
            v16h vb;
            const _Float16* vr;
#pragma unroll
            for (int t = 0; t < 4; ++t) {
                vr = Vb + (size_t)(t * 16 + lm) * Sc + kb + half * 16; // contiguous
#pragma unroll
                for (int e = 0; e < 16; ++e) vb[e] = vr[e];
                v8f* op = (t == 0) ? &o0 : (t == 1) ? &o1 : (t == 2) ? &o2 : &o3;
                *op = __builtin_amdgcn_wmma_f32_16x16x32_f16(false, pa, false, vb,
                                                             (short)0, *op, false, false);
            }
        }
    }

    // ---- normalize + store ctx as f16 [B*S, 256] ---------------------------
    const int b_ = bh >> 2;
    const int h_ = bh & (Hc - 1);
#pragma unroll
    for (int r = 0; r < 8; ++r) {
        int   qrow = qbase + r + half * 8;
        float inv  = 1.0f / rsum[r];
        size_t base = ((size_t)(b_ * Sc + qrow)) * Dc + h_ * HDc + lm;
        C[base + 0]  = (_Float16)(o0[r] * inv);
        C[base + 16] = (_Float16)(o1[r] * inv);
        C[base + 32] = (_Float16)(o2[r] * inv);
        C[base + 48] = (_Float16)(o3[r] * inv);
    }
}

// ---------------------------------------------------------------------------
// Kernel 4: output projection  out = ctx @ Wo^T + bo + b_extra  (f32 out)
// ---------------------------------------------------------------------------
__global__ void out_proj_kernel(const _Float16* __restrict__ Cx,
                                const _Float16* __restrict__ Wo16,
                                const float* __restrict__ bo,
                                const float* __restrict__ bx,
                                float* __restrict__ out)
{
    const int lane = threadIdx.x;
    const int half = lane >> 4;
    const int lm   = lane & 15;
    const int mt   = blockIdx.x;   // 512 row tiles
    const int nt   = blockIdx.y;   // 16 col tiles

    v8f acc = {};
    for (int k0 = 0; k0 < Dc; k0 += 32) {
        v16h a;
        const _Float16* ar = Cx + (size_t)(mt * 16 + lm) * Dc + k0 + half * 16;
#pragma unroll
        for (int e = 0; e < 8; ++e) {
            a[e]     = ar[e];
            a[e + 8] = ar[8 + e];
        }
        v16h b;
        const _Float16* wr = Wo16 + (size_t)(nt * 16 + lm) * Dc + k0 + half * 16;
#pragma unroll
        for (int e = 0; e < 16; ++e) b[e] = wr[e];
        acc = __builtin_amdgcn_wmma_f32_16x16x32_f16(false, a, false, b,
                                                     (short)0, acc, false, false);
    }

    const int n = nt * 16 + lm;
    const float bias = bo[n] + bx[n];
#pragma unroll
    for (int r = 0; r < 8; ++r) {
        int row = mt * 16 + r + half * 8;
        out[(size_t)row * Dc + n] = acc[r] + bias;
    }
}

// ---------------------------------------------------------------------------
extern "C" void kernel_launch(void* const* d_in, const int* in_sizes, int n_in,
                              void* d_out, int out_size, void* d_ws, size_t ws_size,
                              hipStream_t stream)
{
    const float* x  = (const float*)d_in[0];
    const float* G  = (const float*)d_in[1];
    const float* Wq = (const float*)d_in[2];
    const float* Wk = (const float*)d_in[3];
    const float* Wv = (const float*)d_in[4];
    const float* Wo = (const float*)d_in[5];
    const float* bo = (const float*)d_in[6];
    const float* bx = (const float*)d_in[7];
    float* out = (float*)d_out;

    _Float16* ws   = (_Float16*)d_ws;
    _Float16* W16  = ws + OFF_W16;
    _Float16* Q16  = ws + OFF_Q16;
    _Float16* K16  = ws + OFF_K16;
    _Float16* Vt16 = ws + OFF_VT16;
    _Float16* C16  = ws + OFF_C16;

    // 1) weights -> f16 (Wq pre-scaled by 1/8)
    wconv_kernel<<<dim3(4 * 65536 / 256), dim3(256), 0, stream>>>(Wq, Wk, Wv, Wo, W16);

    // 2) QKV projections (WMMA), Q/K -> [B,H,S,HD], V -> [B,H,HD,S]
    qkv_proj_kernel<<<dim3(BSc / 16, Dc / 16, 3), dim3(32), 0, stream>>>(
        x, W16, Q16, K16, Vt16);

    // 3) flash attention with multiplicative G mask (WMMA + online softmax)
    flash_attn_kernel<<<dim3(Bc * Hc * (Sc / 16)), dim3(32), 0, stream>>>(
        Q16, K16, Vt16, G, C16);

    // 4) output projection + biases (WMMA, f32 out)
    out_proj_kernel<<<dim3(BSc / 16, Dc / 16), dim3(32), 0, stream>>>(
        C16, W16 + (size_t)3 * Dc * Dc, bo, bx, out);
}
